// LSTMModel_26517128085514
// MI455X (gfx1250) — compile-verified, matrix operand order checked by reading, not run on hardware
//
#include <hip/hip_runtime.h>
#include <hip/hip_bf16.h>

// ---------------------------------------------------------------------------
// LSTM(B=1024,T=256,U=512) + MLP head for MI455X (gfx1250):
//  - batch-tiled persistent kernel, one WG per 16 batch rows (recurrence local)
//  - h@Wh via v_wmma_f32_16x16x32_bf16; Wh pre-packed into B-fragment layout
//  - Wh (2MB bf16) stays L2-resident; fragments stream L2 -> VGPR each step as
//    global_load_b128 (an integer-offset asm barrier defeats LICM w/o losing
//    the global address space, so loads track LOADcnt only, not DScnt)
//  - c stays in VGPRs; h ping-pongs through LDS (1 barrier per timestep)
// ---------------------------------------------------------------------------

typedef __attribute__((ext_vector_type(16))) __bf16 v16bf;
typedef __attribute__((ext_vector_type(8)))  float  v8f;

#define B_   1024
#define T_   256
#define U_   512
#define G_   2048      // 4*U
#define D_   64
#define P_   32
#define BT   16        // batch rows per workgroup
#define SPAD 520       // padded row stride (elems) for h LDS tile

__device__ __forceinline__ unsigned short f2bf(float x) {
    unsigned u = __float_as_uint(x);
    u += 0x7FFFu + ((u >> 16) & 1u);          // round-to-nearest-even
    return (unsigned short)(u >> 16);
}
__device__ __forceinline__ float bf2f(unsigned short s) {
    return __uint_as_float(((unsigned)s) << 16);
}
__device__ __forceinline__ float sigf(float x)  { return 1.0f / (1.0f + __expf(-x)); }
__device__ __forceinline__ float tanhx(float x) { return 2.0f / (1.0f + __expf(-2.0f * x)) - 1.0f; }

// ---------------------------------------------------------------------------
// Pack Wh [512 x 2048] f32 row-major -> bf16 B-fragments (32B/lane contiguous).
// Fragment (ntg, kk): cols [ntg*16,+16), K rows [kk*32,+32).
//   lanes 0-15  : n = lane,    kc {0..7} then {16..23}
//   lanes 16-31 : n = lane-16, kc {8..15} then {24..31}
// ---------------------------------------------------------------------------
__global__ void pack_wh(const float* __restrict__ Wh, unsigned short* __restrict__ packed) {
    int pe = blockIdx.x * blockDim.x + threadIdx.x;
    if (pe >= U_ * G_) return;
    int e    = pe & 15;
    int lane = (pe >> 4) & 31;
    int kk   = (pe >> 9) & 15;
    int ntg  = pe >> 13;                 // 0..127
    int nl   = lane & 15;
    int hi   = lane >> 4;
    int kc   = (e < 8) ? ((hi ? 8 : 0) + e) : ((hi ? 24 : 16) + (e - 8));
    int k    = kk * 32 + kc;
    int col  = ntg * 16 + nl;
    packed[pe] = f2bf(Wh[k * G_ + col]);
}

// prices [B,T] -> xT [T,B] so per-step staging loads are contiguous
__global__ void transpose_prices(const float* __restrict__ prices, float* __restrict__ xT) {
    int idx = blockIdx.x * blockDim.x + threadIdx.x;
    if (idx >= B_ * T_) return;
    int t = idx / B_, b = idx - t * B_;
    xT[idx] = prices[b * T_ + t];
}

// ---------------------------------------------------------------------------
// Persistent LSTM: 64 WGs x 256 threads (8 waves). Wave w owns hidden units
// [w*64,(w+1)*64): gates i,f,g,o for those units are wave-local, c in VGPRs.
// h is the only cross-wave state: bf16, double-buffered in LDS.
// ---------------------------------------------------------------------------
__global__ __launch_bounds__(256) void lstm_persist(
    const unsigned short* __restrict__ packedWh,  // bf16 B-fragments
    const float* __restrict__ xT,                 // [T,B]
    const float* __restrict__ Wx,                 // [2048]
    const float* __restrict__ bias,               // [2048]
    const float* __restrict__ Wd, const float* __restrict__ bd,
    const float* __restrict__ Wp, const float* __restrict__ bp,
    float* __restrict__ out)                      // [B,32]
{
    __shared__ __align__(16) unsigned short sHbf[2][BT][SPAD]; // ping-pong h
    __shared__ float sPrice[2][BT];
    __shared__ float sX[BT][D_ + 4];

    const int tid  = threadIdx.x;
    const int lane = tid & 31;
    const int w    = tid >> 5;        // wave 0..7
    const int hi   = lane >> 4;       // half-wave
    const int nl   = lane & 15;
    const int b0   = blockIdx.x * BT;
    const int u0   = w * 64;          // this wave's hidden-unit block

    // gate-bias scalars for this lane's columns, held in registers for all T
    float wxg[4][4], bg_[4][4];       // [j][gate]
    #pragma unroll
    for (int j = 0; j < 4; ++j) {
        const int col = u0 + j * 16 + nl;
        #pragma unroll
        for (int q = 0; q < 4; ++q) {
            wxg[j][q] = Wx[q * U_ + col];
            bg_[j][q] = bias[q * U_ + col];
        }
    }

    float c_reg[4][8];
    #pragma unroll
    for (int j = 0; j < 4; ++j)
        #pragma unroll
        for (int v = 0; v < 8; ++v) c_reg[j][v] = 0.f;

    // init: zero h buffer 0, stage prices for t=0
    for (int i = tid; i < BT * SPAD; i += 256) (&sHbf[0][0][0])[i] = 0;
    if (tid < BT) sPrice[0][tid] = xT[b0 + tid];
    __syncthreads();

    // Opaque (always-zero) byte offset: re-defined by empty asm each timestep
    // so the weight loads are not loop-invariant (no hoist -> no spill), while
    // packedWh keeps its global address space (global_load_b128, LOADcnt only).
    unsigned int tweak = 0;

    for (int t = 0; t < T_; ++t) {
        const int cur = t & 1, nxt = cur ^ 1;

        asm volatile("" : "+s"(tweak));

        float pr[8];
        #pragma unroll
        for (int v = 0; v < 8; ++v) pr[v] = sPrice[cur][v + 8 * hi];
        if (tid < BT && t + 1 < T_) sPrice[nxt][tid] = xT[(t + 1) * B_ + b0 + tid];

        #pragma unroll
        for (int j = 0; j < 4; ++j) {
            v8f acc[4] = {};                                       // i,f,g,o
            // per-lane fragment base for (wave,j); q/kk become immediates
            const char* base = (const char*)packedWh + tweak
                             + (w * 4 + j) * 16384 + lane * 32;

            for (int kk = 0; kk < 16; ++kk) {
                union { v16bf v; uint4 q[2]; } A;
                const unsigned short* ap = &sHbf[cur][nl][kk * 32 + hi * 8];
                A.q[0] = *(const uint4*)(ap);
                A.q[1] = *(const uint4*)(ap + 16);
                __builtin_prefetch(base + (kk + 1) * 1024, 0, 3);  // next K-chunk -> WGP$

                #pragma unroll
                for (int q = 0; q < 4; ++q) {
                    union { v16bf v; uint4 u[2]; } Bf;
                    const uint4* bptr = (const uint4*)(base + q * 524288 + kk * 1024);
                    Bf.u[0] = bptr[0];
                    Bf.u[1] = bptr[1];
                    acc[q] = __builtin_amdgcn_wmma_f32_16x16x32_bf16(
                        false, A.v, false, Bf.v, (short)0, acc[q], false, false);
                }
            }

            // gates + state update; write new h into the other buffer
            #pragma unroll
            for (int v = 0; v < 8; ++v) {
                float zi = acc[0][v] + pr[v] * wxg[j][0] + bg_[j][0];
                float zf = acc[1][v] + pr[v] * wxg[j][1] + bg_[j][1];
                float zg = acc[2][v] + pr[v] * wxg[j][2] + bg_[j][2];
                float zo = acc[3][v] + pr[v] * wxg[j][3] + bg_[j][3];
                float ig = sigf(zi), fg = sigf(zf), gg = tanhx(zg), og = sigf(zo);
                float cn = fg * c_reg[j][v] + ig * gg;
                c_reg[j][v] = cn;
                sHbf[nxt][v + 8 * hi][u0 + j * 16 + nl] = f2bf(og * tanhx(cn));
            }
        }
        __syncthreads();    // new h + next prices visible before next step
    }

    // final h lives in buffer (T_ & 1) == 0
    // head, part 1: x = relu(h @ Wd + bd) -> [16,64]
    for (int idx = tid; idx < BT * D_; idx += 256) {
        int m = idx >> 6, d = idx & 63;
        float s = bd[d];
        for (int u = 0; u < U_; ++u)
            s += bf2f(sHbf[0][m][u]) * Wd[u * D_ + d];
        sX[m][d] = s > 0.f ? s : 0.f;
    }
    __syncthreads();
    // head, part 2: out = x @ Wp + bp -> [16,32]
    for (int idx = tid; idx < BT * P_; idx += 256) {
        int m = idx >> 5, p = idx & 31;
        float s = bp[p];
        for (int d = 0; d < D_; ++d)
            s += sX[m][d] * Wp[d * P_ + p];
        out[(b0 + m) * P_ + p] = s;
    }
}

extern "C" void kernel_launch(void* const* d_in, const int* in_sizes, int n_in,
                              void* d_out, int out_size, void* d_ws, size_t ws_size,
                              hipStream_t stream) {
    (void)in_sizes; (void)n_in; (void)out_size; (void)ws_size;
    const float* prices = (const float*)d_in[0];
    const float* Wx     = (const float*)d_in[1];
    const float* Wh     = (const float*)d_in[2];
    const float* bias   = (const float*)d_in[3];
    const float* Wd     = (const float*)d_in[4];
    const float* bd     = (const float*)d_in[5];
    const float* Wp     = (const float*)d_in[6];
    const float* bp     = (const float*)d_in[7];
    float* out = (float*)d_out;

    unsigned short* packedWh = (unsigned short*)d_ws;                               // 2 MB
    float* xT = (float*)((char*)d_ws + (size_t)U_ * G_ * sizeof(unsigned short));   // 1 MB

    hipLaunchKernelGGL(pack_wh, dim3((U_ * G_ + 255) / 256), dim3(256), 0, stream,
                       Wh, packedWh);
    hipLaunchKernelGGL(transpose_prices, dim3((B_ * T_ + 255) / 256), dim3(256), 0, stream,
                       prices, xT);
    hipLaunchKernelGGL(lstm_persist, dim3(B_ / BT), dim3(256), 0, stream,
                       packedWh, xT, Wx, bias, Wd, bd, Wp, bp, out);
}
